// CachedCausalDiT_81870666597076
// MI455X (gfx1250) — matrix operand order repaired and checked by reading, not compiled, not run on hardware
//
#include <hip/hip_runtime.h>
#include <hip/hip_bf16.h>
#include <math.h>

typedef __attribute__((ext_vector_type(16))) _Float16 v16h;
typedef __attribute__((ext_vector_type(8)))  float    v8f;

#define D_      320
#define NH_     20
#define DH_     16
#define TOKS_   65
#define TF_     20
#define B_      2
#define S_      1300
#define ROWS_   (B_*S_)        /* 2600 */
#define NIMG_   (B_*TF_)       /* 40  */
#define CH_     160            /* D/2 */
#define PIX_    576            /* 24*24 */
#define HP_     8
#define NP_     64             /* patches per frame */
#define PD_     1440           /* CH*3*3 */
#define MODW_   1920           /* 6*D */
#define LOGC_   8.517193191416238f /* ln(5000) */

__device__ __forceinline__ float siluf(float x) { return x / (1.0f + __expf(-x)); }

// ---------------------------------------------------------------------------
// Generic GEMM:  C[M,N] = A[M,K] (f32) * W[K,N] (f32) + bias[N]
// f16 WMMA 16x16x32. Block = 4 waves; block tile 32(M) x 64(N);
// each wave holds two 16x16 accumulators sharing one B fragment.
// Staging uses GLOBAL_LOAD_ASYNC_TO_LDS_B128 (ASYNCcnt-tracked DMA into LDS):
// tiles land in LDS as raw f32; f32->f16 conversion happens at fragment build.
// Out-of-range rows/cols are clamped, never zero-filled: garbage tile rows/
// cols only feed C rows/cols that the guarded epilogue discards, and clamping
// keeps every 16B async chunk inside the source buffers.
// Requires K % 32 == 0 (true for all call sites).
// ---------------------------------------------------------------------------
__global__ void __launch_bounds__(128)
gemm_f16wmma(const float* __restrict__ A, const float* __restrict__ W,
             const float* __restrict__ bias, float* __restrict__ C,
             int M, int N, int K)
{
    __shared__ float sAf[32 * 32];     // M-major: sAf[m*32 + k]
    __shared__ float sBf[32 * 64];     // K-major: sBf[k*64 + n]
    const int tid  = threadIdx.x;
    const int lane = tid & 31;
    const int wv   = tid >> 5;
    const int m0   = blockIdx.y * 32;
    const int n0   = blockIdx.x * 64;

    v8f acc0 = {}, acc1 = {};
    const int arow = lane & 15;               // A fragment M index (sub-tile)
    const int akb  = (lane >> 4) << 3;        // A fragment K base: 0 or 8
    const int brow = (wv << 4) + (lane & 15); // B fragment N index in 64-wide tile
    const int bkb  = (lane >> 4) << 3;        // B fragment K base: 0 or 8

    // ---- async staging chunk assignments (16B float4 granularity) ----
    // A tile: 256 chunks; this thread owns chunks {tid, tid+128}
    const int a0r = tid >> 3;                 // row 0..15
    const int a0c = (tid & 7) * 4;            // k offset 0,4,..,28
    const int a1r = a0r + 16;                 // row 16..31
    const int a0g = (m0 + a0r < M) ? (m0 + a0r) : (M - 1);
    const int a1g = (m0 + a1r < M) ? (m0 + a1r) : (M - 1);
    const unsigned ldsA0 = (unsigned)(size_t)&sAf[a0r * 32 + a0c];
    const unsigned ldsA1 = (unsigned)(size_t)&sAf[a1r * 32 + a0c];
    // B tile: 512 chunks; this thread owns chunks {tid + 128*p : p=0..3}
    const int bk  = tid >> 4;                 // k 0..7 (+8 per p)
    const int bn4 = (tid & 15) * 4;           // n offset 0,4,..,60
    int bcol = n0 + bn4;
    if (bcol > N - 4) bcol = N - 4;           // keep 16B chunk in-bounds
    const unsigned ldsB0 = (unsigned)(size_t)&sBf[bk * 64 + bn4];

    for (int k0 = 0; k0 < K; k0 += 32) {
        const float* pa0 = A + (size_t)a0g * K + k0 + a0c;
        const float* pa1 = A + (size_t)a1g * K + k0 + a0c;
        asm volatile("global_load_async_to_lds_b128 %0, %1, off"
                     :: "v"(ldsA0), "v"((unsigned long long)(size_t)pa0) : "memory");
        asm volatile("global_load_async_to_lds_b128 %0, %1, off"
                     :: "v"(ldsA1), "v"((unsigned long long)(size_t)pa1) : "memory");
        #pragma unroll
        for (int p = 0; p < 4; ++p) {
            const float* pb = W + (size_t)(k0 + bk + 8 * p) * N + bcol;
            unsigned lb = ldsB0 + (unsigned)(8 * p * 64 * sizeof(float));
            asm volatile("global_load_async_to_lds_b128 %0, %1, off"
                         :: "v"(lb), "v"((unsigned long long)(size_t)pb) : "memory");
        }
        asm volatile("s_wait_asynccnt 0x0" ::: "memory");
        __syncthreads();

        v16h af0, af1, bf;
        #pragma unroll
        for (int j = 0; j < 8; ++j) {
            int ka = (j < 4) ? (akb + 2 * j) : (16 + akb + 2 * (j - 4));
            af0[2 * j]     = (_Float16)sAf[arow * 32 + ka];
            af0[2 * j + 1] = (_Float16)sAf[arow * 32 + ka + 1];
            af1[2 * j]     = (_Float16)sAf[(arow + 16) * 32 + ka];
            af1[2 * j + 1] = (_Float16)sAf[(arow + 16) * 32 + ka + 1];
            int kb = (j < 4) ? (bkb + 2 * j) : (16 + bkb + 2 * (j - 4));
            bf[2 * j]     = (_Float16)sBf[kb * 64 + brow];
            bf[2 * j + 1] = (_Float16)sBf[(kb + 1) * 64 + brow];
        }
        acc0 = __builtin_amdgcn_wmma_f32_16x16x32_f16(false, af0, false, bf,
                                                      (short)0, acc0, false, false);
        acc1 = __builtin_amdgcn_wmma_f32_16x16x32_f16(false, af1, false, bf,
                                                      (short)0, acc1, false, false);
        __syncthreads();
    }
    const int cl = lane & 15;
    const int rb = (lane >> 4) << 3;
    const int col = n0 + (wv << 4) + cl;
    if (col < N) {
        float b = bias ? bias[col] : 0.0f;
        #pragma unroll
        for (int i = 0; i < 8; ++i) {
            int row0 = m0 + rb + i;
            int row1 = row0 + 16;
            if (row0 < M) C[(size_t)row0 * N + col] = acc0[i] + b;
            if (row1 < M) C[(size_t)row1 * N + col] = acc1[i] + b;
        }
    }
}

// ---------------------------------------------------------------------------
// Implicit-GEMM 5x5 conv (pad 2, stride 1) + bias + SiLU on the WMMA pipe.
//   M = NIMG*576 output pixels, N = COUT, K = CIN*25 (zero-padded to mult 32)
// A tile staging performs the im2col gather (can't be expressed as DMA);
// epilogue scatters to NCHW. conv weights w[co][ci][ky][kx] are N-major.
// ---------------------------------------------------------------------------
__global__ void __launch_bounds__(128)
conv5_gemm_wmma(const float* __restrict__ in, const float* __restrict__ w,
                const float* __restrict__ bias, float* __restrict__ out,
                int CIN, int COUT)
{
    const int Kr = CIN * 25;
    const int Kp = (Kr + 31) & ~31;
    const int M  = NIMG_ * PIX_;
    const int N  = COUT;
    __shared__ _Float16 sA[32 * 32];
    __shared__ _Float16 sB[64 * 32];
    const int tid  = threadIdx.x;
    const int lane = tid & 31;
    const int wv   = tid >> 5;
    const int m0   = blockIdx.y * 32;
    const int n0   = blockIdx.x * 64;

    v8f acc0 = {}, acc1 = {};
    const int arow = lane & 15;
    const int akb  = (lane >> 4) << 3;
    const int brow = (wv << 4) + (lane & 15);
    const int bkb  = (lane >> 4) << 3;

    const int ar  = tid >> 2;
    const int ac0 = (tid & 3) * 8;
    const int prow = m0 + ar;               // output pixel index (M % 32 == 0)
    const int pimg = prow / PIX_;
    const int ppix = prow - pimg * PIX_;
    const int py   = ppix / 24;
    const int px   = ppix - py * 24;
    const int bn   = tid & 63;
    const int bkp  = tid >> 6;
    const int bcol = n0 + bn;
    const int bcl  = bcol < N ? bcol : (N - 1);
    const bool bval = bcol < N;

    for (int k0 = 0; k0 < Kp; k0 += 32) {
        // A tile: im2col gather, 8 taps per thread
        float va[8];
        #pragma unroll
        for (int j = 0; j < 8; ++j) {
            int k = k0 + ac0 + j;
            float v = 0.0f;
            if (k < Kr) {
                int ci  = k / 25;
                int rem = k - ci * 25;
                int ky  = rem / 5;
                int kx  = rem - ky * 5;
                int yy  = py + ky - 2;
                int xx  = px + kx - 2;
                if ((unsigned)yy < 24u && (unsigned)xx < 24u)
                    v = in[((size_t)pimg * CIN + ci) * PIX_ + yy * 24 + xx];
            }
            va[j] = v;
        }
        // B tile: weight gather, N-major / K-contiguous in memory
        float vb[16];
        #pragma unroll
        for (int t = 0; t < 16; ++t) {
            int k = k0 + bkp + 2 * t;
            float v = (bval && k < Kr) ? w[(size_t)bcl * Kr + k] : 0.0f;
            vb[t] = v;
        }
        #pragma unroll
        for (int j = 0; j < 8; ++j)
            sA[ar * 32 + ac0 + j] = (_Float16)va[j];
        #pragma unroll
        for (int t = 0; t < 16; ++t)
            sB[bn * 32 + bkp + 2 * t] = (_Float16)vb[t];
        __syncthreads();

        v16h af0, af1, bf;
        #pragma unroll
        for (int j = 0; j < 8; ++j) {
            int ka = (j < 4) ? (akb + 2 * j) : (16 + akb + 2 * (j - 4));
            af0[2 * j]     = sA[arow * 32 + ka];
            af0[2 * j + 1] = sA[arow * 32 + ka + 1];
            af1[2 * j]     = sA[(arow + 16) * 32 + ka];
            af1[2 * j + 1] = sA[(arow + 16) * 32 + ka + 1];
            int kb = (j < 4) ? (bkb + 2 * j) : (16 + bkb + 2 * (j - 4));
            bf[2 * j]     = sB[brow * 32 + kb];
            bf[2 * j + 1] = sB[brow * 32 + kb + 1];
        }
        acc0 = __builtin_amdgcn_wmma_f32_16x16x32_f16(false, af0, false, bf,
                                                      (short)0, acc0, false, false);
        acc1 = __builtin_amdgcn_wmma_f32_16x16x32_f16(false, af1, false, bf,
                                                      (short)0, acc1, false, false);
        __syncthreads();
    }
    const int cl = lane & 15;
    const int rb = (lane >> 4) << 3;
    const int col = n0 + (wv << 4) + cl;
    if (col < N) {
        float b = bias[col];
        #pragma unroll
        for (int i = 0; i < 8; ++i) {
            int row0 = m0 + rb + i;
            int row1 = row0 + 16;
            {
                int n = row0 / PIX_, pix = row0 - n * PIX_;
                out[((size_t)n * COUT + col) * PIX_ + pix] = siluf(acc0[i] + b);
            }
            {
                int n = row1 / PIX_, pix = row1 - n * PIX_;
                out[((size_t)n * COUT + col) * PIX_ + pix] = siluf(acc1[i] + b);
            }
        }
    }
}

// ---------------------------------------------------------------------------
// cond = time_pe[ts] @ tmix_w + tmix_b + action_emb[a]   (one block per row)
// ---------------------------------------------------------------------------
__global__ void __launch_bounds__(320)
cond_embed(const float* __restrict__ time_pe, const int* __restrict__ ts_idx,
           const int* __restrict__ actions, const float* __restrict__ tmix_w,
           const float* __restrict__ tmix_b, const float* __restrict__ aemb,
           float* __restrict__ cond)
{
    __shared__ float pe[D_];
    int row = blockIdx.x;
    int d = threadIdx.x;
    int ts = ts_idx[row];
    int a  = actions[row];
    pe[d] = time_pe[(size_t)ts * D_ + d];
    __syncthreads();
    float acc = tmix_b[d];
    for (int k = 0; k < D_; ++k) acc += pe[k] * tmix_w[(size_t)k * D_ + d];
    cond[(size_t)row * D_ + d] = acc + aemb[(size_t)a * D_ + d];
}

__global__ void elt_silu(const float* __restrict__ in, float* __restrict__ out, int n)
{
    int i = blockIdx.x * blockDim.x + threadIdx.x;
    if (i < n) out[i] = siluf(in[i]);
}

// GroupNorm, 32 groups over 160ch x 24x24; one block per (image, group)
__global__ void __launch_bounds__(256)
groupnorm32(const float* __restrict__ in, const float* __restrict__ w,
            const float* __restrict__ b, float* __restrict__ out)
{
    const int CG = CH_ / 32;          // 5
    int n = blockIdx.x >> 5;
    int g = blockIdx.x & 31;
    const int cnt = CG * PIX_;        // 2880
    const float* base = in + ((size_t)n * CH_ + g * CG) * PIX_;
    float s = 0.f, ss = 0.f;
    for (int i = threadIdx.x; i < cnt; i += 256) { float v = base[i]; s += v; ss += v * v; }
    __shared__ float rs[256], rq[256];
    rs[threadIdx.x] = s; rq[threadIdx.x] = ss;
    __syncthreads();
    for (int off = 128; off > 0; off >>= 1) {
        if (threadIdx.x < off) { rs[threadIdx.x] += rs[threadIdx.x + off];
                                 rq[threadIdx.x] += rq[threadIdx.x + off]; }
        __syncthreads();
    }
    float mean = rs[0] / cnt;
    float var  = rq[0] / cnt - mean * mean;
    float inv  = rsqrtf(var + 1e-5f);
    float* ob = out + ((size_t)n * CH_ + g * CG) * PIX_;
    for (int i = threadIdx.x; i < cnt; i += 256) {
        int c = g * CG + i / PIX_;
        ob[i] = (base[i] - mean) * inv * w[c] + b[c];
    }
}

// (n,c,24,24) -> patches (n, 64, 1440) with col = c*9 + ph*3 + pw
__global__ void patchify(const float* __restrict__ act, float* __restrict__ patches)
{
    int idx = blockIdx.x * blockDim.x + threadIdx.x;
    if (idx >= NIMG_ * NP_ * PD_) return;
    int j  = idx % PD_;
    int p  = (idx / PD_) % NP_;
    int n  = idx / (PD_ * NP_);
    int c  = j / 9; int r9 = j % 9; int ph = r9 / 3; int pw = r9 % 3;
    int hp = p / HP_; int wp = p % HP_;
    patches[idx] = act[((size_t)n * CH_ + c) * PIX_ + (hp * 3 + ph) * 24 + (wp * 3 + pw)];
}

// insert register token (row 0 of each frame)
__global__ void assemble_tokens(const float* __restrict__ tokens,
                                const float* __restrict__ regs, float* __restrict__ z)
{
    int idx = blockIdx.x * blockDim.x + threadIdx.x;
    if (idx >= ROWS_ * D_) return;
    int d = idx % D_;
    int s = (idx / D_) % S_;
    int b = idx / (D_ * S_);
    int t = s / TOKS_; int r = s % TOKS_;
    z[idx] = (r == 0) ? regs[d]
                      : tokens[((size_t)(b * TF_ + t) * NP_ + (r - 1)) * D_ + d];
}

__global__ void rope_build(float* __restrict__ sins, float* __restrict__ coss)
{
    int idx = blockIdx.x * blockDim.x + threadIdx.x;
    if (idx >= S_ * DH_) return;
    int d = idx % DH_; int pos = idx / DH_;
    float fr  = __expf(-LOGC_ * (float)(d & ~1) / (float)DH_);
    float ang = (float)pos * fr;
    sins[idx] = sinf(ang);
    coss[idx] = cosf(ang);
}

// h = rmsnorm(x, nw) * (1 + sg) + mu   with per-frame (b, f) modulation rows
__global__ void __launch_bounds__(320)
rms_modulate(const float* __restrict__ x, const float* __restrict__ nw,
             const float* __restrict__ mod, int modStride, int muOff, int sgOff,
             float* __restrict__ out)
{
    __shared__ float red[320];
    int row = blockIdx.x;        // b*S + s
    int d = threadIdx.x;
    int b = row / S_;
    int s = row - b * S_;
    int f = s / TOKS_;
    float v = x[(size_t)row * D_ + d];
    red[d] = v * v;
    __syncthreads();
    if (d < 160) red[d] += red[d + 160]; __syncthreads();
    if (d < 80)  red[d] += red[d + 80];  __syncthreads();
    if (d < 40)  red[d] += red[d + 40];  __syncthreads();
    if (d < 20)  red[d] += red[d + 20];  __syncthreads();
    if (d < 10)  red[d] += red[d + 10];  __syncthreads();
    if (d < 5)   red[d] += red[d + 5];   __syncthreads();
    if (d == 0)  red[0] = red[0] + red[1] + red[2] + red[3] + red[4];
    __syncthreads();
    float rms = rsqrtf(red[0] / (float)D_ + 1e-6f);
    const float* mrow = mod + (size_t)(b * TF_ + f) * modStride;
    out[(size_t)row * D_ + d] = v * rms * nw[d] * (1.0f + mrow[sgOff + d]) + mrow[muOff + d];
}

// per-head rmsnorm(lnq/lnk) + RoPE; split qkv into (b*NH, S, DH) layouts
__global__ void qk_prep(const float* __restrict__ qkv, const float* __restrict__ lnq,
                        const float* __restrict__ lnk, const float* __restrict__ sins,
                        const float* __restrict__ coss, float* __restrict__ qh,
                        float* __restrict__ kh, float* __restrict__ vh)
{
    int idx = blockIdx.x * blockDim.x + threadIdx.x;
    if (idx >= ROWS_ * NH_) return;
    int h  = idx % NH_;
    int rs = idx / NH_;            // b*S + s
    int b  = rs / S_;
    int s  = rs - b * S_;
    const float* qr = qkv + (size_t)rs * (3 * D_) + h * DH_;
    const float* kr = qr + D_;
    const float* vr = qr + 2 * D_;
    float q[DH_], k[DH_];
    float sq = 0.f, sk = 0.f;
    #pragma unroll
    for (int i = 0; i < DH_; ++i) { q[i] = qr[i]; k[i] = kr[i]; sq += q[i]*q[i]; sk += k[i]*k[i]; }
    float rq = rsqrtf(sq / (float)DH_ + 1e-6f);
    float rk = rsqrtf(sk / (float)DH_ + 1e-6f);
    #pragma unroll
    for (int i = 0; i < DH_; ++i) { q[i] *= rq * lnq[i]; k[i] *= rk * lnk[i]; }
    size_t base = ((size_t)(b * NH_ + h) * S_ + s) * DH_;
    const float* sn = sins + (size_t)s * DH_;
    const float* cs = coss + (size_t)s * DH_;
    #pragma unroll
    for (int i = 0; i < DH_ / 2; ++i) {
        float c = cs[2 * i], si = sn[2 * i];
        qh[base + 2*i]     = c * q[2*i]     - si * q[2*i + 1];
        qh[base + 2*i + 1] = c * q[2*i + 1] + si * q[2*i];
        kh[base + 2*i]     = c * k[2*i]     - si * k[2*i + 1];
        kh[base + 2*i + 1] = c * k[2*i + 1] + si * k[2*i];
        vh[base + 2*i]     = vr[2*i];
        vh[base + 2*i + 1] = vr[2*i + 1];
    }
}

// one wave per (b, head, query): streaming softmax over frame-causal prefix
__global__ void __launch_bounds__(256)
attn_flash(const float* __restrict__ qh, const float* __restrict__ kh,
           const float* __restrict__ vh, float* __restrict__ outb)
{
    const int lane = threadIdx.x & 31;
    const int wv   = threadIdx.x >> 5;
    int item = blockIdx.x * 8 + wv;
    const int total = B_ * NH_ * S_;
    if (item >= total) return;
    int q = item % S_;
    int h = (item / S_) % NH_;
    int b = item / (S_ * NH_);
    int kmax = (q / TOKS_ + 1) * TOKS_;     // frame-causal: >= 65 >= wave width
    size_t hb = (size_t)(b * NH_ + h) * S_;
    const float* qv = qh + (hb + q) * DH_;
    float qr[DH_];
    #pragma unroll
    for (int i = 0; i < DH_; ++i) qr[i] = qv[i];
    float m = -3.0e38f, l = 0.f;
    float acc[DH_];
    #pragma unroll
    for (int i = 0; i < DH_; ++i) acc[i] = 0.f;
    for (int kk = lane; kk < kmax; kk += 32) {
        const float* kv = kh + (hb + kk) * DH_;
        const float* vv = vh + (hb + kk) * DH_;
        float sc = 0.f;
        #pragma unroll
        for (int i = 0; i < DH_; ++i) sc += qr[i] * kv[i];
        float nm = fmaxf(m, sc);
        float sa = __expf(m - nm);
        float p  = __expf(sc - nm);
        l = l * sa + p;
        #pragma unroll
        for (int i = 0; i < DH_; ++i) acc[i] = acc[i] * sa + p * vv[i];
        m = nm;
    }
    #pragma unroll
    for (int off = 16; off >= 1; off >>= 1) {
        float om = __shfl_xor(m, off, 32);
        float ol = __shfl_xor(l, off, 32);
        float oa[DH_];
        #pragma unroll
        for (int i = 0; i < DH_; ++i) oa[i] = __shfl_xor(acc[i], off, 32);
        float nm = fmaxf(m, om);
        float sa = __expf(m - nm);
        float sb = __expf(om - nm);
        l = l * sa + ol * sb;
        #pragma unroll
        for (int i = 0; i < DH_; ++i) acc[i] = acc[i] * sa + oa[i] * sb;
        m = nm;
    }
    if (lane == 0) {
        float inv = 1.0f / l;
        float* orow = outb + (size_t)(b * S_ + q) * D_ + h * DH_;
        #pragma unroll
        for (int i = 0; i < DH_; ++i) orow[i] = acc[i] * inv;
    }
}

// z += c[b,frame] * val  (adaLN gate)
__global__ void gate_add(float* __restrict__ z, const float* __restrict__ val,
                         const float* __restrict__ mod, int cOff, int n)
{
    int idx = blockIdx.x * blockDim.x + threadIdx.x;
    if (idx >= n) return;
    int d = idx % D_;
    int row = idx / D_;
    int b = row / S_;
    int s = row - b * S_;
    int f = s / TOKS_;
    z[idx] += mod[(size_t)(b * TF_ + f) * MODW_ + cOff + d] * val[idx];
}

__global__ void swiglu_mul(float* __restrict__ u, const float* __restrict__ g, int n)
{
    int i = blockIdx.x * blockDim.x + threadIdx.x;
    if (i < n) u[i] *= siluf(g[i]);
}

// drop register tokens: (b, t*65 + 1 + p) -> (b*TF+t)*64 + p
__global__ void gather_final(const float* __restrict__ hF, float* __restrict__ yin)
{
    int idx = blockIdx.x * blockDim.x + threadIdx.x;
    if (idx >= NIMG_ * NP_ * D_) return;
    int d  = idx % D_;
    int r  = idx / D_;
    int p  = r % NP_;
    int bt = r / NP_;
    int t  = bt % TF_;
    int b  = bt / TF_;
    yin[idx] = hF[((size_t)(b * S_ + t * TOKS_ + 1 + p)) * D_ + d];
}

// (n, 64, 27)=(hp,wp | p,q,c) -> (n, 3, 24, 24)
__global__ void unpatch_scatter(const float* __restrict__ yout, float* __restrict__ out)
{
    int idx = blockIdx.x * blockDim.x + threadIdx.x;
    if (idx >= NIMG_ * 3 * PIX_) return;
    int x = idx % 24;
    int y = (idx / 24) % 24;
    int c = (idx / PIX_) % 3;
    int n = idx / (PIX_ * 3);
    int hp = y / 3, pr = y % 3, wp = x / 3, pc = x % 3;
    out[idx] = yout[((size_t)n * NP_ + hp * HP_ + wp) * 27 + pr * 9 + pc * 3 + c];
}

// ---------------------------------------------------------------------------
static inline void launch_gemm(const float* A, const float* W, const float* bias,
                               float* C, int M, int N, int K, hipStream_t st)
{
    dim3 g((N + 63) / 64, (M + 31) / 32), b(128);
    gemm_f16wmma<<<g, b, 0, st>>>(A, W, bias, C, M, N, K);
}

#define PF(i) ((const float*)d_in[(i)])

extern "C" void kernel_launch(void* const* d_in, const int* in_sizes, int n_in,
                              void* d_out, int out_size, void* d_ws, size_t ws_size,
                              hipStream_t stream)
{
    (void)in_sizes; (void)n_in; (void)out_size; (void)ws_size;
    // jax tree_flatten order: x, actions, ts_idx, then params dict (keys sorted):
    // action_emb, blocks[0..7]{down_b,down_w,gate_b,gate_w,lnk_w,lnq_w,mod_b,mod_w,
    //   n1_w,n2_w,o_b,o_w,qkv_b,qkv_w,up_b,up_w}, conv1_b, conv1_w, conv2_b, conv2_w,
    //   emb_b, emb_w, fmod_b, fmod_w, fnorm_w, gn1_b, gn1_w, gn2_b, gn2_w, registers,
    //   time_pe, tmix_b, tmix_w, unpatch_b, unpatch_w
    const float* x_in       = PF(0);
    const int*   actions    = (const int*)d_in[1];
    const int*   ts_idx     = (const int*)d_in[2];
    const float* action_emb = PF(3);
    const float* conv1_b = PF(132);
    const float* conv1_w = PF(133);
    const float* conv2_b = PF(134);
    const float* conv2_w = PF(135);
    const float* emb_b   = PF(136);
    const float* emb_w   = PF(137);
    const float* fmod_b  = PF(138);
    const float* fmod_w  = PF(139);
    const float* fnorm_w = PF(140);
    const float* gn1_b   = PF(141);
    const float* gn1_w   = PF(142);
    const float* gn2_b   = PF(143);
    const float* gn2_w   = PF(144);
    const float* registers = PF(145);
    const float* time_pe = PF(146);
    const float* tmix_b  = PF(147);
    const float* tmix_w  = PF(148);
    const float* unpatch_b = PF(149);
    const float* unpatch_w = PF(150);

    float* WS = (float*)d_ws;
    size_t off = 0;
    auto carve = [&](size_t nf) { float* p = WS + off; off += nf; return p; };
    float* bufA   = carve(3686400);   // conv ping / patches; reused as MLP u
    float* bufB   = carve(3686400);   // conv pong; tail reused by MLP g
    float* tokens = carve(819200);
    float* zbuf   = carve(832000);
    float* hbuf   = carve(832000);
    float* qkvb   = carve(2496000);
    float* qhb    = carve(832000);
    float* khb    = carve(832000);
    float* vhb    = carve(832000);
    float* attnb  = carve(832000);
    float* projb  = carve(832000);    // attn proj, then MLP down output
    float* condb  = carve(12800);
    float* scb    = carve(12800);
    float* modb   = carve(76800);
    float* fmodb  = carve(25600);
    float* sinb   = carve(20800);
    float* cosb   = carve(20800);
    float* yinb   = carve(819200);
    float* youtb  = carve(69120);
    float* ubuf = bufA;                // 3,328,000 floats
    float* gbuf = bufA + 3328000;      // spills into bufB region (free by then)

    dim3 b256(256);

    // conditioning
    cond_embed<<<dim3(NIMG_), dim3(320), 0, stream>>>(time_pe, ts_idx, actions,
                                                      tmix_w, tmix_b, action_emb, condb);
    elt_silu<<<dim3((NIMG_ * D_ + 255) / 256), b256, 0, stream>>>(condb, scb, NIMG_ * D_);

    // patch embedding: both convs run as implicit GEMM on the WMMA pipe
    {
        dim3 gc1((CH_ + 63) / 64, (NIMG_ * PIX_ + 31) / 32), bc(128);
        conv5_gemm_wmma<<<gc1, bc, 0, stream>>>(x_in, conv1_w, conv1_b, bufA, 3, CH_);
        groupnorm32<<<dim3(NIMG_ * 32), b256, 0, stream>>>(bufA, gn1_w, gn1_b, bufB);
        conv5_gemm_wmma<<<gc1, bc, 0, stream>>>(bufB, conv2_w, conv2_b, bufA, CH_, CH_);
        groupnorm32<<<dim3(NIMG_ * 32), b256, 0, stream>>>(bufA, gn2_w, gn2_b, bufB);
    }
    patchify<<<dim3((NIMG_ * NP_ * PD_ + 255) / 256), b256, 0, stream>>>(bufB, bufA);
    launch_gemm(bufA, emb_w, emb_b, tokens, NIMG_ * NP_, D_, PD_, stream);
    assemble_tokens<<<dim3((ROWS_ * D_ + 255) / 256), b256, 0, stream>>>(tokens, registers, zbuf);
    rope_build<<<dim3((S_ * DH_ + 255) / 256), b256, 0, stream>>>(sinb, cosb);

    for (int blk = 0; blk < 8; ++blk) {
        const int base = 4 + blk * 16;
        const float* down_b = PF(base + 0);
        const float* down_w = PF(base + 1);
        const float* gate_b = PF(base + 2);
        const float* gate_w = PF(base + 3);
        const float* lnk_w  = PF(base + 4);
        const float* lnq_w  = PF(base + 5);
        const float* mod_b  = PF(base + 6);
        const float* mod_w  = PF(base + 7);
        const float* n1_w   = PF(base + 8);
        const float* n2_w   = PF(base + 9);
        const float* o_b    = PF(base + 10);
        const float* o_w    = PF(base + 11);
        const float* qkv_b  = PF(base + 12);
        const float* qkv_w  = PF(base + 13);
        const float* up_b   = PF(base + 14);
        const float* up_w   = PF(base + 15);

        launch_gemm(scb, mod_w, mod_b, modb, NIMG_, MODW_, D_, stream);
        rms_modulate<<<dim3(ROWS_), dim3(320), 0, stream>>>(
            zbuf, n1_w, modb, MODW_, 0, D_, hbuf);
        launch_gemm(hbuf, qkv_w, qkv_b, qkvb, ROWS_, 3 * D_, D_, stream);
        qk_prep<<<dim3((ROWS_ * NH_ + 255) / 256), b256, 0, stream>>>(
            qkvb, lnq_w, lnk_w, sinb, cosb, qhb, khb, vhb);
        attn_flash<<<dim3((B_ * NH_ * S_ + 7) / 8), b256, 0, stream>>>(qhb, khb, vhb, attnb);
        launch_gemm(attnb, o_w, o_b, projb, ROWS_, D_, D_, stream);
        gate_add<<<dim3((ROWS_ * D_ + 255) / 256), b256, 0, stream>>>(
            zbuf, projb, modb, 2 * D_, ROWS_ * D_);
        rms_modulate<<<dim3(ROWS_), dim3(320), 0, stream>>>(
            zbuf, n2_w, modb, MODW_, 3 * D_, 4 * D_, hbuf);
        launch_gemm(hbuf, up_w, up_b, ubuf, ROWS_, 4 * D_, D_, stream);
        launch_gemm(hbuf, gate_w, gate_b, gbuf, ROWS_, 4 * D_, D_, stream);
        swiglu_mul<<<dim3((ROWS_ * 4 * D_ + 255) / 256), b256, 0, stream>>>(
            ubuf, gbuf, ROWS_ * 4 * D_);
        launch_gemm(ubuf, down_w, down_b, projb, ROWS_, D_, 4 * D_, stream);
        gate_add<<<dim3((ROWS_ * D_ + 255) / 256), b256, 0, stream>>>(
            zbuf, projb, modb, 5 * D_, ROWS_ * D_);
    }

    // final modulate + unpatch
    launch_gemm(scb, fmod_w, fmod_b, fmodb, NIMG_, 2 * D_, D_, stream);
    rms_modulate<<<dim3(ROWS_), dim3(320), 0, stream>>>(
        zbuf, fnorm_w, fmodb, 2 * D_, 0, D_, hbuf);
    gather_final<<<dim3((NIMG_ * NP_ * D_ + 255) / 256), b256, 0, stream>>>(hbuf, yinb);
    launch_gemm(yinb, unpatch_w, unpatch_b, youtb, NIMG_ * NP_, 27, D_, stream);
    unpatch_scatter<<<dim3((NIMG_ * 3 * PIX_ + 255) / 256), b256, 0, stream>>>(
        youtb, (float*)d_out);
}
